// Transformer_6751688590029
// MI455X (gfx1250) — compile-verified, matrix operand order checked by reading, not run on hardware
//
#include <hip/hip_runtime.h>
#include <hip/hip_bf16.h>

#define DEV __device__ __forceinline__

typedef __attribute__((ext_vector_type(16))) __bf16        v16bf;
typedef __attribute__((ext_vector_type(8)))  float         v8f;
typedef __attribute__((ext_vector_type(8)))  unsigned int  v8u;
typedef __attribute__((ext_vector_type(4)))  unsigned int  v4u;
typedef __attribute__((ext_vector_type(4)))  unsigned int  u32x4;
typedef __attribute__((ext_vector_type(8)))  int           i32x8;
typedef __attribute__((ext_vector_type(4)))  int           i32x4;

constexpr int B_ = 4, L_ = 1024, D_ = 512, H_ = 8;
constexpr int BL = B_ * L_;                 // 4096
constexpr float NEGBIG = -1.0e9f;
#define FNEG_INF (-3.0e38f)

#if defined(__has_builtin)
#if __has_builtin(__builtin_amdgcn_tensor_load_to_lds) && \
    __has_builtin(__builtin_amdgcn_s_wait_tensorcnt)
#define HAVE_TDM 1
#endif
#endif

// ---------------- workspace layout (bytes) ----------------
constexpr size_t SZ_TOKBF = (size_t)BL * D_ * 2;     // 4 MiB  (bf16 [4096,512])
constexpr size_t SZ_WBF   = (size_t)D_ * D_ * 2;     // 512 KiB
constexpr size_t OFF_QBF   = 0;
constexpr size_t OFF_KVBF  = OFF_QBF  + SZ_TOKBF;
constexpr size_t OFF_WQT   = OFF_KVBF + SZ_TOKBF;    // W^T (bf16, [N][K])
constexpr size_t OFF_WKT   = OFF_WQT  + SZ_WBF;
constexpr size_t OFF_WVT   = OFF_WKT  + SZ_WBF;
constexpr size_t OFF_WOT   = OFF_WVT  + SZ_WBF;
constexpr size_t OFF_QPB   = OFF_WOT  + SZ_WBF;      // projected Q (pad-masked)
constexpr size_t OFF_KB    = OFF_QPB  + SZ_TOKBF;    // projected K
constexpr size_t OFF_VT    = OFF_KB   + SZ_TOKBF;    // projected V, TRANSPOSED [B][D][L]
constexpr size_t OFF_OUTHB = OFF_VT   + SZ_TOKBF;    // attn^T V, head-merged bf16
constexpr size_t OFF_SCAL  = OFF_OUTHB+ SZ_TOKBF;
constexpr size_t SZ_SCAL   = 1024;
constexpr size_t OFF_LOG   = OFF_SCAL + SZ_SCAL;     // f32 [B,H,L,L] = 134 MB
constexpr size_t SZ_LOG    = (size_t)B_*H_*L_*L_*4;
constexpr size_t OFF_EBF   = OFF_LOG + SZ_LOG;       // bf16 [B,H,L,L] transposed = 67 MB
// scal: 0 maxL | 1 maxT | 2..33 m_bh | 34..65 sum_bh | 66..97 s_bh | 98..101 rowcnt | 102 maxAttn

// ---------------- helpers ----------------
DEV unsigned short f2bf(float f) {                    // f32 -> bf16 (RNE)
  unsigned u = __float_as_uint(f);
  u += 0x7FFFu + ((u >> 16) & 1u);
  return (unsigned short)(u >> 16);
}

DEV void atomicMaxF(float* addr, float val) {         // ordered-int float max
  if (val >= 0.0f) atomicMax((int*)addr, __float_as_int(val));
  else             atomicMin((unsigned int*)addr, __float_as_uint(val));
}

DEV float waveRedMax(float v) {
#pragma unroll
  for (int o = 16; o; o >>= 1) v = fmaxf(v, __shfl_xor(v, o, 32));
  return v;
}
DEV float waveRedSum(float v) {
#pragma unroll
  for (int o = 16; o; o >>= 1) v += __shfl_xor(v, o, 32);
  return v;
}

// 16x32 bf16 fragment, contraction-contiguous memory: exactly two b128 loads.
// Layout (ISA 7.12.2): lane&15 -> M/N index; VGPR v pair-base = (v>>2)*16 + (lane>>4)*8 + (v&3)*2.
// For v=0..3 that is 8 consecutive ushorts at +((lane>>4)*8); v=4..7 the same at +16.
DEV v16bf load_frag16(const unsigned short* base, int major, int ld, int k0, int lane) {
  const unsigned short* p = base + (size_t)major * ld + k0 + ((lane >> 4) << 3);
  v4u lo = *(const v4u*)(p);
  v4u hi = *(const v4u*)(p + 16);
  v8u r = __builtin_shufflevector(lo, hi, 0, 1, 2, 3, 4, 5, 6, 7);
  return __builtin_bit_cast(v16bf, r);
}

DEV v8f wmma_bf16(v16bf a, v16bf b, v8f c) {
  return __builtin_amdgcn_wmma_f32_16x16x32_bf16(false, a, false, b, (short)0, c, false, false);
}

// Stage an n-float row from global into LDS via the Tensor Data Mover.
DEV void stage_vec_lds(const float* gsrc, float* sdst, int n) {
#ifdef HAVE_TDM
  if (threadIdx.x < 32) {
    unsigned lds_off = (unsigned)(unsigned long long)
        (__attribute__((address_space(3))) float*)sdst;
    unsigned long long ga = (unsigned long long)gsrc;
    // D# group0: count=1 | lds_addr | global_addr[56:0] | type=2 (bits 127:126)
    u32x4 g0 = { 1u, lds_off, (unsigned)ga,
                 ((unsigned)(ga >> 32) & 0x01FFFFFFu) | 0x80000000u };
    // D# group1: data_size=4B, 1-D tile: tensor_dim0=tile_dim0=n, dim1=1, stride0=n
    unsigned un = (unsigned)n;
    i32x8 g1 = { (int)0x20000,                 // data_size=2 (4 bytes) at bits 17:16
                 (int)((un & 0xFFFFu) << 16),  // tensor_dim0[15:0] at bits 63:48
                 (int)((un >> 16) | (1u << 16)), // tensor_dim0[31:16] | tensor_dim1 lo=1
                 (int)(un << 16),              // tile_dim0 at bits 127:112
                 1,                            // tile_dim1 = 1
                 (int)un,                      // tensor_dim0_stride[31:0]
                 0, 0 };
    i32x4 z4 = { 0, 0, 0, 0 };
#if __clang_major__ >= 23
    i32x8 z8 = { 0, 0, 0, 0, 0, 0, 0, 0 };
    __builtin_amdgcn_tensor_load_to_lds(g0, g1, z4, z4, z8, 0);
#else
    __builtin_amdgcn_tensor_load_to_lds(g0, g1, z4, z4, 0);
#endif
    __builtin_amdgcn_s_wait_tensorcnt(0);
  }
#else
  for (int t = threadIdx.x; t < n; t += blockDim.x) sdst[t] = gsrc[t];
#endif
  __syncthreads();
}

// ---------------- kernels ----------------
__global__ void init_scalars_kernel(float* scal) {
  int i = threadIdx.x;
  if (i < 103) {
    float v = 0.0f;
    if (i < 34 || i == 102) v = FNEG_INF;
    scal[i] = v;
  }
}

// tokens -> bf16 ; weights -> bf16 TRANSPOSED ([N][K]) so GEMM B-frags are K-contiguous.
__global__ __launch_bounds__(256) void prep_kernel(
    const float* q, const float* kv, const float* wq, const float* wk,
    const float* wv, const float* wo, unsigned short* qb, unsigned short* kvb,
    unsigned short* wqT, unsigned short* wkT, unsigned short* wvT, unsigned short* woT) {
  int stride = gridDim.x * blockDim.x;
  int i0 = blockIdx.x * blockDim.x + threadIdx.x;
  const int NTOK = BL * D_;
  for (int i = i0; i < NTOK; i += stride) { qb[i] = f2bf(q[i]); kvb[i] = f2bf(kv[i]); }
  const int NW = D_ * D_;
  for (int i = i0; i < NW; i += stride) {
    int n = i >> 9, k = i & (D_ - 1);
    size_t src = (size_t)k * D_ + n;          // transpose through L2
    wqT[i] = f2bf(wq[src]); wkT[i] = f2bf(wk[src]);
    wvT[i] = f2bf(wv[src]); woT[i] = f2bf(wo[src]);
  }
}

// Q/K/V projections: [4096,512] x Wt + bias. 32x32 per wave (2x2 WMMA tiling).
// Q rows pad-masked; V written transposed [B][D][L] for the attn^T V GEMM.
__global__ __launch_bounds__(256) void proj_kernel(
    const unsigned short* qbf, const unsigned short* kvbf,
    const unsigned short* wqT, const unsigned short* wkT, const unsigned short* wvT,
    const float* bq, const float* bk, const float* bv, const float* protok,
    unsigned short* qpb, unsigned short* kout, unsigned short* vt) {
  int which = blockIdx.y;                                   // 0:Q 1:K 2:V
  const unsigned short* A  = (which == 0) ? qbf : kvbf;
  const unsigned short* WT = (which == 0) ? wqT : (which == 1) ? wkT : wvT;
  const float* bias        = (which == 0) ? bq  : (which == 1) ? bk  : bv;

  __shared__ float sbias[D_];
  stage_vec_lds(bias, sbias, D_);                           // TDM -> LDS

  int wave = threadIdx.x >> 5, lane = threadIdx.x & 31;
  int tid = blockIdx.x * 8 + wave;                          // 2048 tiles (128 x 16)
  int tm = tid >> 4, tn = tid & 15;
  int r0 = tm * 32 + (lane & 15);
  int c0 = tn * 32 + (lane & 15);

  v8f acc[2][2] = {};
#pragma unroll 4
  for (int k0 = 0; k0 < D_; k0 += 32) {
    __builtin_prefetch(A + (size_t)r0 * D_ + k0 + 128, 0, 1);
    v16bf fa0 = load_frag16(A,  r0,      D_, k0, lane);
    v16bf fa1 = load_frag16(A,  r0 + 16, D_, k0, lane);
    v16bf fb0 = load_frag16(WT, c0,      D_, k0, lane);
    v16bf fb1 = load_frag16(WT, c0 + 16, D_, k0, lane);
    acc[0][0] = wmma_bf16(fa0, fb0, acc[0][0]);
    acc[0][1] = wmma_bf16(fa0, fb1, acc[0][1]);
    acc[1][0] = wmma_bf16(fa1, fb0, acc[1][0]);
    acc[1][1] = wmma_bf16(fa1, fb1, acc[1][1]);
  }
#pragma unroll
  for (int ti = 0; ti < 2; ++ti)
#pragma unroll
    for (int tj = 0; tj < 2; ++tj) {
      int cc = tn * 32 + tj * 16 + (lane & 15);
      float bn = sbias[cc];
#pragma unroll
      for (int r = 0; r < 8; ++r) {
        int m = tm * 32 + ti * 16 + r + ((lane >> 4) << 3);
        float val = acc[ti][tj][r] + bn;
        if (which == 0 && protok[m] == 0.0f) val = 0.0f;    // padding_to_zeros on Q
        if (which == 2)                                     // V transposed: [B][D][L]
          vt[((size_t)(m >> 10) * D_ + cc) * L_ + (m & (L_ - 1))] = f2bf(val);
        else if (which == 0)
          qpb[(size_t)m * D_ + cc] = f2bf(val);
        else
          kout[(size_t)m * D_ + cc] = f2bf(val);
      }
    }
}

// logits[b,h] = Q_h K_h^T (64 contraction), 32x32 per wave, fused global max.
__global__ __launch_bounds__(256) void qk_kernel(
    const unsigned short* qpb, const unsigned short* kbf, float* logits, float* scal) {
  int wave = threadIdx.x >> 5, lane = threadIdx.x & 31;
  int tid = blockIdx.x * 8 + wave;                          // 32768 tiles
  int bh = tid >> 10;                                       // 32x32 tiles per (b,h)
  int t  = tid & 1023;
  int tm = t >> 5, tn = t & 31;
  int b = bh >> 3, h = bh & 7;
  const unsigned short* Qb = qpb + (size_t)b * L_ * D_ + h * 64;
  const unsigned short* Kb = kbf + (size_t)b * L_ * D_ + h * 64;
  int r0 = tm * 32 + (lane & 15);
  int c0 = tn * 32 + (lane & 15);

  v8f acc[2][2] = {};
#pragma unroll
  for (int k0 = 0; k0 < 64; k0 += 32) {
    v16bf fa0 = load_frag16(Qb, r0,      D_, k0, lane);
    v16bf fa1 = load_frag16(Qb, r0 + 16, D_, k0, lane);
    v16bf fb0 = load_frag16(Kb, c0,      D_, k0, lane);
    v16bf fb1 = load_frag16(Kb, c0 + 16, D_, k0, lane);
    acc[0][0] = wmma_bf16(fa0, fb0, acc[0][0]);
    acc[0][1] = wmma_bf16(fa0, fb1, acc[0][1]);
    acc[1][0] = wmma_bf16(fa1, fb0, acc[1][0]);
    acc[1][1] = wmma_bf16(fa1, fb1, acc[1][1]);
  }
  float* out = logits + (size_t)bh * (L_ * L_);
  float m = FNEG_INF;
#pragma unroll
  for (int ti = 0; ti < 2; ++ti)
#pragma unroll
    for (int tj = 0; tj < 2; ++tj) {
      int cc = tn * 32 + tj * 16 + (lane & 15);
#pragma unroll
      for (int r = 0; r < 8; ++r) {
        int mm = tm * 32 + ti * 16 + r + ((lane >> 4) << 3);
        out[(size_t)mm * L_ + cc] = acc[ti][tj][r];
        m = fmaxf(m, acc[ti][tj][r]);
      }
    }
  m = waveRedMax(m);
  __shared__ float sm[8];
  if ((threadIdx.x & 31) == 0) sm[threadIdx.x >> 5] = m;
  __syncthreads();
  if (threadIdx.x < 32) {
    float v = (threadIdx.x < 8) ? sm[threadIdx.x] : FNEG_INF;
    v = waveRedMax(v);
    if (threadIdx.x == 0) atomicMaxF(&scal[0], v);          // global max(logits)
  }
}

__global__ __launch_bounds__(256) void statmax_kernel(const float* hb, const float* pi, float* scal) {
  const size_t N = (size_t)B_ * L_ * L_;
  size_t stride = (size_t)gridDim.x * blockDim.x;
  float m = FNEG_INF;
  for (size_t i = blockIdx.x * (size_t)blockDim.x + threadIdx.x; i < N; i += stride)
    m = fmaxf(m, fmaxf(hb[i], pi[i]));
  m = waveRedMax(m);
  __shared__ float sm[8];
  if ((threadIdx.x & 31) == 0) sm[threadIdx.x >> 5] = m;
  __syncthreads();
  if (threadIdx.x < 32) {
    float v = (threadIdx.x < 8) ? sm[threadIdx.x] : FNEG_INF;
    v = waveRedMax(v);
    if (threadIdx.x == 0) atomicMaxF(&scal[1], v);          // global max(table)
  }
}

// c = attw*logit/maxL + stat/maxT + pad_row*NEG, in place, fused per-(b,h) max.
__global__ __launch_bounds__(256) void combine_kernel(
    float* logits, const float* hb, const float* pi, const float* protok,
    const float* attw, const float* augw, float* scal) {
  int bh = blockIdx.y, b = bh >> 3, h = bh & 7;
  const float* stat = ((h < 4) ? hb : pi) + (size_t)b * L_ * L_;  // jnp.repeat(H/2) pattern
  float aw = attw[0] / scal[0];
  float gw = 1.0f / scal[1];            // aug weight cancels in table/max(aug*table)
  (void)augw;
  float* Lg = logits + (size_t)bh * (L_ * L_);
  const float* pk = protok + b * L_;
  float m = FNEG_INF;
  int stride = gridDim.x * blockDim.x;
  for (int i = blockIdx.x * blockDim.x + threadIdx.x; i < L_ * L_; i += stride) {
    int qrow = i >> 10;
    float c = aw * Lg[i] + gw * stat[i] + ((pk[qrow] == 0.0f) ? NEGBIG : 0.0f);
    Lg[i] = c;
    m = fmaxf(m, c);
  }
  m = waveRedMax(m);
  __shared__ float sm[8];
  if ((threadIdx.x & 31) == 0) sm[threadIdx.x >> 5] = m;
  __syncthreads();
  if (threadIdx.x < 32) {
    float v = (threadIdx.x < 8) ? sm[threadIdx.x] : FNEG_INF;
    v = waveRedMax(v);
    if (threadIdx.x == 0) atomicMaxF(&scal[2 + bh], v);     // per-(b,h) softmax max
  }
}

// e = exp(c - m_bh), stored TRANSPOSED (kv-major) bf16 for attn^T @ V; fused sum.
__global__ __launch_bounds__(256) void softexp_kernel(
    const float* logits, unsigned short* Ebf, float* scal) {
  int bh = blockIdx.y;
  float mb = scal[2 + bh];
  const float* Lg = logits + (size_t)bh * (L_ * L_);
  unsigned short* E = Ebf + (size_t)bh * (L_ * L_);
  float s = 0.0f;
  int stride = gridDim.x * blockDim.x;
  for (int i = blockIdx.x * blockDim.x + threadIdx.x; i < L_ * L_; i += stride) {
    float e = __expf(Lg[i] - mb);
    int qrow = i >> 10, kcol = i & (L_ - 1);
    E[(size_t)kcol * L_ + qrow] = f2bf(e);                  // transpose absorbed by 192MB L2
    s += e;
  }
  s = waveRedSum(s);
  __shared__ float sm[8];
  if ((threadIdx.x & 31) == 0) sm[threadIdx.x >> 5] = s;
  __syncthreads();
  if (threadIdx.x < 32) {
    float v = (threadIdx.x < 8) ? sm[threadIdx.x] : 0.0f;
    v = waveRedSum(v);
    if (threadIdx.x == 0) atomicAdd(&scal[34 + bh], v);
  }
}

// attn = e/sum * cnt / maxAttn; maxAttn = max_bh(cnt_b/sum_bh) since max(e)==1.
__global__ void finalize_kernel(const float* protok, float* scal) {
  __shared__ float cnt[4];
  __shared__ float rv[32];
  int t = threadIdx.x;
  if (t < 4) cnt[t] = 0.0f;
  __syncthreads();
  for (int i = t; i < BL; i += blockDim.x)
    if (protok[i] != 0.0f) atomicAdd(&cnt[i >> 10], 1.0f);
  __syncthreads();
  if (t < 32) rv[t] = cnt[t >> 3] / scal[34 + t];
  __syncthreads();
  if (t == 0) {
    float m = FNEG_INF;
    for (int j = 0; j < 32; ++j) m = fmaxf(m, rv[j]);
    scal[102] = m;
  }
  __syncthreads();
  if (t < 32) scal[66 + t] = rv[t] / scal[102];
  if (t < 4)  scal[98 + t] = cnt[t];
}

// out_h[b,h,kv,d] = s_bh * sum_q E^T[kv,q] * V^T[d,q] -> bf16 [B,L,D] head-merged.
__global__ __launch_bounds__(256) void av_kernel(
    const unsigned short* Ebf, const unsigned short* vt, const float* scal,
    unsigned short* outh) {
  int wave = threadIdx.x >> 5, lane = threadIdx.x & 31;
  int tid = blockIdx.x * 8 + wave;                          // 2048 tiles (32bh x 32 x 2)
  int bh = tid >> 6;
  int t  = tid & 63;
  int tm = t >> 1, tn = t & 1;
  int b = bh >> 3, h = bh & 7;
  const unsigned short* A   = Ebf + (size_t)bh * (L_ * L_);             // [kv][q]
  const unsigned short* VTb = vt + ((size_t)b * D_ + h * 64) * L_;      // [d][q]
  int r0 = tm * 32 + (lane & 15);                           // kv index
  int c0 = tn * 32 + (lane & 15);                           // d in [0,64)

  v8f acc[2][2] = {};
#pragma unroll 4
  for (int k0 = 0; k0 < L_; k0 += 32) {
    __builtin_prefetch(A + (size_t)r0 * L_ + k0 + 128, 0, 1);
    v16bf fa0 = load_frag16(A,   r0,      L_, k0, lane);
    v16bf fa1 = load_frag16(A,   r0 + 16, L_, k0, lane);
    v16bf fb0 = load_frag16(VTb, c0,      L_, k0, lane);
    v16bf fb1 = load_frag16(VTb, c0 + 16, L_, k0, lane);
    acc[0][0] = wmma_bf16(fa0, fb0, acc[0][0]);
    acc[0][1] = wmma_bf16(fa0, fb1, acc[0][1]);
    acc[1][0] = wmma_bf16(fa1, fb0, acc[1][0]);
    acc[1][1] = wmma_bf16(fa1, fb1, acc[1][1]);
  }
  float s = scal[66 + bh];
#pragma unroll
  for (int ti = 0; ti < 2; ++ti)
#pragma unroll
    for (int tj = 0; tj < 2; ++tj) {
      int cc = tn * 32 + tj * 16 + (lane & 15);
#pragma unroll
      for (int r = 0; r < 8; ++r) {
        int kk = tm * 32 + ti * 16 + r + ((lane >> 4) << 3);
        outh[((size_t)(b * L_ + kk)) * D_ + h * 64 + cc] = f2bf(acc[ti][tj][r] * s);
      }
    }
}

// final projection: [4096,512] x Wo^T + bo, pad-masked, f32 out.
__global__ __launch_bounds__(256) void out_kernel(
    const unsigned short* outh, const unsigned short* woT, const float* bo,
    const float* protok, float* out) {
  __shared__ float sbias[D_];
  stage_vec_lds(bo, sbias, D_);                             // TDM -> LDS

  int wave = threadIdx.x >> 5, lane = threadIdx.x & 31;
  int tid = blockIdx.x * 8 + wave;                          // 2048 tiles
  int tm = tid >> 4, tn = tid & 15;
  int r0 = tm * 32 + (lane & 15);
  int c0 = tn * 32 + (lane & 15);

  v8f acc[2][2] = {};
#pragma unroll 4
  for (int k0 = 0; k0 < D_; k0 += 32) {
    __builtin_prefetch(outh + (size_t)r0 * D_ + k0 + 128, 0, 1);
    v16bf fa0 = load_frag16(outh, r0,      D_, k0, lane);
    v16bf fa1 = load_frag16(outh, r0 + 16, D_, k0, lane);
    v16bf fb0 = load_frag16(woT,  c0,      D_, k0, lane);
    v16bf fb1 = load_frag16(woT,  c0 + 16, D_, k0, lane);
    acc[0][0] = wmma_bf16(fa0, fb0, acc[0][0]);
    acc[0][1] = wmma_bf16(fa0, fb1, acc[0][1]);
    acc[1][0] = wmma_bf16(fa1, fb0, acc[1][0]);
    acc[1][1] = wmma_bf16(fa1, fb1, acc[1][1]);
  }
#pragma unroll
  for (int ti = 0; ti < 2; ++ti)
#pragma unroll
    for (int tj = 0; tj < 2; ++tj) {
      int cc = tn * 32 + tj * 16 + (lane & 15);
      float bn = sbias[cc];
#pragma unroll
      for (int r = 0; r < 8; ++r) {
        int m = tm * 32 + ti * 16 + r + ((lane >> 4) << 3);
        float val = acc[ti][tj][r] + bn;
        if (protok[m] == 0.0f) val = 0.0f;
        out[(size_t)m * D_ + cc] = val;
      }
    }
}

// ---------------- launch ----------------
extern "C" void kernel_launch(void* const* d_in, const int* in_sizes, int n_in,
                              void* d_out, int out_size, void* d_ws, size_t ws_size,
                              hipStream_t stream) {
  const float* q      = (const float*)d_in[0];
  const float* kv     = (const float*)d_in[1];
  const float* protok = (const float*)d_in[2];
  const float* hb     = (const float*)d_in[3];
  const float* pi     = (const float*)d_in[4];
  // d_in[5] cross_mask == pad[b,q] broadcast; derived from protok instead.
  const float* wq = (const float*)d_in[6];
  const float* bq = (const float*)d_in[7];
  const float* wk = (const float*)d_in[8];
  const float* bk = (const float*)d_in[9];
  const float* wv = (const float*)d_in[10];
  const float* bv = (const float*)d_in[11];
  const float* wo = (const float*)d_in[12];
  const float* bo = (const float*)d_in[13];
  const float* attw = (const float*)d_in[14];
  const float* augw = (const float*)d_in[15];

  char* ws = (char*)d_ws;
  unsigned short* qb   = (unsigned short*)(ws + OFF_QBF);
  unsigned short* kvb  = (unsigned short*)(ws + OFF_KVBF);
  unsigned short* wqT  = (unsigned short*)(ws + OFF_WQT);
  unsigned short* wkT  = (unsigned short*)(ws + OFF_WKT);
  unsigned short* wvT  = (unsigned short*)(ws + OFF_WVT);
  unsigned short* woT  = (unsigned short*)(ws + OFF_WOT);
  unsigned short* qpb  = (unsigned short*)(ws + OFF_QPB);
  unsigned short* kbf  = (unsigned short*)(ws + OFF_KB);
  unsigned short* vt   = (unsigned short*)(ws + OFF_VT);
  unsigned short* outh = (unsigned short*)(ws + OFF_OUTHB);
  float*          scal = (float*)(ws + OFF_SCAL);
  float*          logg = (float*)(ws + OFF_LOG);
  unsigned short* Ebf  = (unsigned short*)(ws + OFF_EBF);
  (void)in_sizes; (void)n_in; (void)out_size; (void)ws_size;

  init_scalars_kernel<<<1, 128, 0, stream>>>(scal);
  prep_kernel<<<2048, 256, 0, stream>>>(q, kv, wq, wk, wv, wo, qb, kvb, wqT, wkT, wvT, woT);
  proj_kernel<<<dim3(256, 3), 256, 0, stream>>>(qb, kvb, wqT, wkT, wvT, bq, bk, bv, protok,
                                                qpb, kbf, vt);
  qk_kernel<<<4096, 256, 0, stream>>>(qpb, kbf, logg, scal);
  statmax_kernel<<<2048, 256, 0, stream>>>(hb, pi, scal);
  combine_kernel<<<dim3(512, 32), 256, 0, stream>>>(logg, hb, pi, protok, attw, augw, scal);
  softexp_kernel<<<dim3(512, 32), 256, 0, stream>>>(logg, Ebf, scal);
  finalize_kernel<<<1, 256, 0, stream>>>(protok, scal);
  av_kernel<<<256, 256, 0, stream>>>(Ebf, vt, scal, outh);
  out_kernel<<<256, 256, 0, stream>>>(outh, woT, bo, protok, (float*)d_out);
}
